// IterBlock_w_Str_18399639896111
// MI455X (gfx1250) — matrix operand order repaired and checked by reading, not compiled, not run on hardware
//
#include <hip/hip_runtime.h>
#include <hip/hip_bf16.h>
#include <cstring>

// ---------------------------------------------------------------------------
// CDNA5 / gfx1250 implementation of one RoseTTAFold-style iteration block.
// All dense contractions run on v_wmma_f32_16x16x32_bf16 (fp32 -> bf16 A/B,
// f32 accumulate).  One generalized strided/batched WMMA GEMM covers every
// linear layer, every attention QK^T / AV (incl. tied attention via a 2-level
// K decomposition done with shift/mask), and the P^2 outer product (2-level
// M/N store decomposition).  Each wave computes a 16x32 C tile (one shared A
// fragment, two B fragments, two WMMAs per 32-wide K block) to halve A-side
// traffic per FLOP.  Operands are staged through float4 (global_load_b128).
// The 3x3 convs are WMMA implicit-GEMM with the same tiling.  LN/softmax/
// graph ops are streaming kernels.  Scratch: bump-allocated from d_ws.
// ---------------------------------------------------------------------------

typedef __attribute__((ext_vector_type(16))) __bf16 v16bf;
typedef __attribute__((ext_vector_type(8)))  float  v8f;

// ------------------------------ WMMA GEMM ----------------------------------
// C[m,n] (+batches b1,b2) = act( alpha * sum_{k2,ki} A * B + bias[n] ) + Cin
// A addr = b1*b1A + b2*b2A + m*rsA + ki*ksA + k2*k2sA          (Ktot = K*K2)
// B addr = b1*b1B + b2*b2B + n*csB + ki*ksB + k2*k2sB
// C addr = b1*b1C + b2*b2C + (m/MiC)*rsC + (m%MiC)*rsC2
//                          + (n/NiC)*csC + (n%NiC)*csC2
// When K2>1, K must be a power of two: k2 = ka>>kshift, ki = ka&kmask.
struct GemmP {
  const float* A; const float* B; const float* bias; const float* Cin; float* C;
  int M, N, K, K2, nb2, MiC, NiC, act, preact, kshift, kmask;
  float alpha;
  long rsA, ksA, k2sA, b1A, b2A;
  long csB, ksB, k2sB, b1B, b2B;
  long rsC, rsC2, csC, csC2, b1C, b2C;
};

__device__ __forceinline__ void gemm_loadB(const GemmP& g, const float* Bp, int col,
                                           int kb, int kg, int Ktot,
                                           bool fastB, bool full, float bv[16]) {
  if (fastB && full) {
    if (col < g.N) {
      const float* bp = Bp + (long)col * g.csB + kb + kg * 16;
      const float4 q0 = *(const float4*)(bp);
      const float4 q1 = *(const float4*)(bp + 4);
      const float4 q2 = *(const float4*)(bp + 8);
      const float4 q3 = *(const float4*)(bp + 12);
      bv[0]=q0.x; bv[1]=q0.y; bv[2]=q0.z;  bv[3]=q0.w;
      bv[4]=q1.x; bv[5]=q1.y; bv[6]=q1.z;  bv[7]=q1.w;
      bv[8]=q2.x; bv[9]=q2.y; bv[10]=q2.z; bv[11]=q2.w;
      bv[12]=q3.x; bv[13]=q3.y; bv[14]=q3.z; bv[15]=q3.w;
    } else {
#pragma unroll
      for (int i = 0; i < 16; ++i) bv[i] = 0.f;
    }
  } else {
#pragma unroll
    for (int i = 0; i < 16; ++i) {
      const int kbn = kb + kg * 16 + i;
      float v = 0.f;
      if (col < g.N && kbn < Ktot) {
        const int k2 = (g.K2 > 1) ? (kbn >> g.kshift) : 0;
        const int ki = (g.K2 > 1) ? (kbn & g.kmask) : kbn;
        v = Bp[(long)col * g.csB + (long)ki * g.ksB + (long)k2 * g.k2sB];
      }
      bv[i] = v;
    }
  }
}

__device__ __forceinline__ void gemm_store(const GemmP& g, v8f acc, int mb, int col,
                                           int b1, int b2) {
  if (col >= g.N) return;
#pragma unroll
  for (int r = 0; r < 8; ++r) {
    const int m = mb + r;
    if (m >= g.M) continue;
    float v = acc[r] * g.alpha;
    if (g.bias) v += g.bias[col];
    if (g.act == 1) v = fmaxf(v, 0.f);
    else if (g.act == 2) v = (v > 0.f) ? v : (__expf(v) - 1.f);
    const long caddr = (long)(m / g.MiC) * g.rsC + (long)(m % g.MiC) * g.rsC2
                     + (long)(col / g.NiC) * g.csC + (long)(col % g.NiC) * g.csC2
                     + (long)b1 * g.b1C + (long)b2 * g.b2C;
    if (g.Cin) v += g.Cin[caddr];
    g.C[caddr] = v;
  }
}

__global__ __launch_bounds__(128) void gemm_wmma_kernel(GemmP g) {
  const int lane = threadIdx.x & 31;
  const int wave = threadIdx.x >> 5;
  const int m0 = blockIdx.x * 32 + (wave >> 1) * 16;
  const int n0 = blockIdx.y * 64 + (wave & 1) * 32;  // wave -> 16x32 C tile
  if (m0 >= g.M || n0 >= g.N) return;                // wave-uniform exit
  int b1 = (int)blockIdx.z, b2 = 0;
  if (g.nb2 > 1) { b2 = b1 % g.nb2; b1 /= g.nb2; }
  const float* Ap = g.A + (long)b1 * g.b1A + (long)b2 * g.b2A;
  const float* Bp = g.B + (long)b1 * g.b1B + (long)b2 * g.b2B;
  const int row  = m0 + (lane & 15);
  const int col0 = n0 + (lane & 15);
  const int col1 = col0 + 16;
  const int kg   = lane >> 4;                        // half-wave (wave32)
  const int Ktot = g.K * g.K2;
  // vectorizable operands: contiguous in K, 16B-aligned leading stride
  const bool fastA = (g.K2 == 1) && (g.ksA == 1) && ((g.rsA & 3) == 0);
  const bool fastB = (g.K2 == 1) && (g.ksB == 1) && ((g.csB & 3) == 0);
  v8f acc0 = {}, acc1 = {};
  for (int kb = 0; kb < Ktot; kb += 32) {
    if (row < g.M)  // speculative prefetch of next A K-slab (global_prefetch_b8)
      __builtin_prefetch(&Ap[(long)row * g.rsA + (long)(kb + 32) * g.ksA], 0, 1);
    const bool full = (kb + 32 <= Ktot);
    float av[16], bv0[16], bv1[16];
    // ---- A 16x32 bf16 fragment: lanes 0-15 = rows, K = kg*16 + i
    if (fastA && full) {
      if (row < g.M) {
        const float* ap = Ap + (long)row * g.rsA + kb + kg * 16;
        const float4 q0 = *(const float4*)(ap);
        const float4 q1 = *(const float4*)(ap + 4);
        const float4 q2 = *(const float4*)(ap + 8);
        const float4 q3 = *(const float4*)(ap + 12);
        av[0]=q0.x; av[1]=q0.y; av[2]=q0.z;  av[3]=q0.w;
        av[4]=q1.x; av[5]=q1.y; av[6]=q1.z;  av[7]=q1.w;
        av[8]=q2.x; av[9]=q2.y; av[10]=q2.z; av[11]=q2.w;
        av[12]=q3.x; av[13]=q3.y; av[14]=q3.z; av[15]=q3.w;
      } else {
#pragma unroll
        for (int i = 0; i < 16; ++i) av[i] = 0.f;
      }
    } else {
#pragma unroll
      for (int i = 0; i < 16; ++i) {
        const int ka = kb + kg * 16 + i;
        float v = 0.f;
        if (row < g.M && ka < Ktot) {
          const int k2 = (g.K2 > 1) ? (ka >> g.kshift) : 0;
          const int ki = (g.K2 > 1) ? (ka & g.kmask) : ka;
          v = Ap[(long)row * g.rsA + (long)ki * g.ksA + (long)k2 * g.k2sA];
        }
        av[i] = v;
      }
    }
    if (g.preact == 2) {
#pragma unroll
      for (int i = 0; i < 16; ++i) av[i] = (av[i] > 0.f) ? av[i] : (__expf(av[i]) - 1.f);
    }
    // ---- two B 32x16 bf16 fragments: col = lane%16 (+16), K = kg*16 + i
    gemm_loadB(g, Bp, col0, kb, kg, Ktot, fastB, full, bv0);
    gemm_loadB(g, Bp, col1, kb, kg, Ktot, fastB, full, bv1);
    v16bf af, bf0, bf1;
#pragma unroll
    for (int i = 0; i < 16; ++i) {
      af[i] = (__bf16)av[i]; bf0[i] = (__bf16)bv0[i]; bf1[i] = (__bf16)bv1[i];
    }
    acc0 = __builtin_amdgcn_wmma_f32_16x16x32_bf16(false, af, false, bf0,
                                                   (short)0, acc0, false, false);
    acc1 = __builtin_amdgcn_wmma_f32_16x16x32_bf16(false, af, false, bf1,
                                                   (short)0, acc1, false, false);
  }
  const int mb = m0 + kg * 8;                        // C: VGPR r -> M = r + 8*kg
  gemm_store(g, acc0, mb, col0, b1, b2);
  gemm_store(g, acc1, mb, col1, b1, b2);
}

// ----------------------- WMMA implicit-GEMM 3x3 conv -----------------------
// Requires Ci == 128 (channels-last input, OIHW weights).  A 32-wide K block
// never crosses a 3x3-tap boundary (kb % 32 == 0, Ci % 32 == 0), so each
// block needs one bounds check and two float4 pairs per half-wave.  Each
// wave computes a 16x32 output tile (A fragment shared by two WMMAs).
struct ConvP {
  const float* in; const float* W; const float* bias; const float* Cin; float* out;
  int H, Wd, Ci, Co, preelu;
};

__device__ __forceinline__ void conv_loadB(const ConvP& c, int col, int cib, int t9,
                                           int kg, float bv[16]) {
  if (col < c.Co) {
    const float* bp = c.W + ((long)col * 128 + cib + kg * 16) * 9 + t9;  // OIHW
#pragma unroll
    for (int i = 0; i < 16; ++i) bv[i] = bp[i * 9];
  } else {
#pragma unroll
    for (int i = 0; i < 16; ++i) bv[i] = 0.f;
  }
}

__device__ __forceinline__ void conv_store(const ConvP& c, v8f acc, int mb, int col, int M) {
  if (col >= c.Co) return;
#pragma unroll
  for (int r = 0; r < 8; ++r) {
    const int m = mb + r;
    if (m >= M) continue;
    float v = acc[r];
    if (c.bias) v += c.bias[col];
    const long a = (long)m * c.Co + col;
    if (c.Cin) v += c.Cin[a];
    c.out[a] = v;
  }
}

__global__ __launch_bounds__(128) void conv3x3_wmma_kernel(ConvP c) {
  const int lane = threadIdx.x & 31;
  const int wave = threadIdx.x >> 5;
  const int M = c.H * c.Wd, Ktot = 9 * 128;
  const int m0 = blockIdx.x * 32 + (wave >> 1) * 16;
  const int n0 = blockIdx.y * 64 + (wave & 1) * 32;
  if (m0 >= M || n0 >= c.Co) return;
  const int row = m0 + (lane & 15);
  const int y = row / c.Wd, x = row % c.Wd;
  const int col0 = n0 + (lane & 15);
  const int col1 = col0 + 16;
  const int kg  = lane >> 4;
  v8f acc0 = {}, acc1 = {};
  for (int kb = 0; kb < Ktot; kb += 32) {
    const int t9  = kb >> 7;          // tap index (Ci == 128)
    const int cib = kb & 127;         // channel base inside tap
    const int yy = y + t9 / 3 - 1, xx = x + (t9 % 3) - 1;
    const bool inb = (row < M) && (yy >= 0) && (yy < c.H) && (xx >= 0) && (xx < c.Wd);
    float av[16], bv0[16], bv1[16];
    if (inb) {
      const float* ap = c.in + ((long)yy * c.Wd + xx) * 128 + cib + kg * 16;
      const float4 q0 = *(const float4*)(ap);
      const float4 q1 = *(const float4*)(ap + 4);
      const float4 q2 = *(const float4*)(ap + 8);
      const float4 q3 = *(const float4*)(ap + 12);
      av[0]=q0.x; av[1]=q0.y; av[2]=q0.z;  av[3]=q0.w;
      av[4]=q1.x; av[5]=q1.y; av[6]=q1.z;  av[7]=q1.w;
      av[8]=q2.x; av[9]=q2.y; av[10]=q2.z; av[11]=q2.w;
      av[12]=q3.x; av[13]=q3.y; av[14]=q3.z; av[15]=q3.w;
    } else {
#pragma unroll
      for (int i = 0; i < 16; ++i) av[i] = 0.f;
    }
    if (c.preelu) {
#pragma unroll
      for (int i = 0; i < 16; ++i) av[i] = (av[i] > 0.f) ? av[i] : (__expf(av[i]) - 1.f);
    }
    conv_loadB(c, col0, cib, t9, kg, bv0);
    conv_loadB(c, col1, cib, t9, kg, bv1);
    v16bf af, bf0, bf1;
#pragma unroll
    for (int i = 0; i < 16; ++i) {
      af[i] = (__bf16)av[i]; bf0[i] = (__bf16)bv0[i]; bf1[i] = (__bf16)bv1[i];
    }
    acc0 = __builtin_amdgcn_wmma_f32_16x16x32_bf16(false, af, false, bf0,
                                                   (short)0, acc0, false, false);
    acc1 = __builtin_amdgcn_wmma_f32_16x16x32_bf16(false, af, false, bf1,
                                                   (short)0, acc1, false, false);
  }
  const int mb = m0 + kg * 8;
  conv_store(c, acc0, mb, col0, M);
  conv_store(c, acc1, mb, col1, M);
}

// ------------------------------ LayerNorm ----------------------------------
__global__ __launch_bounds__(128) void ln_kernel(const float* __restrict__ x,
    const float* __restrict__ g, const float* __restrict__ b,
    float* __restrict__ y, int d, float eps) {
  __shared__ float sred[4];
  const long row = blockIdx.x;
  const float* xr = x + row * (long)d;
  float* yr = y + row * (long)d;
  float s = 0.f;
  for (int i = threadIdx.x; i < d; i += blockDim.x) s += xr[i];
  for (int o = 16; o > 0; o >>= 1) s += __shfl_down(s, o);
  if ((threadIdx.x & 31) == 0) sred[threadIdx.x >> 5] = s;
  __syncthreads();
  const float mean = (sred[0] + sred[1] + sred[2] + sred[3]) / d;
  __syncthreads();
  float v = 0.f;
  for (int i = threadIdx.x; i < d; i += blockDim.x) { float t = xr[i] - mean; v += t * t; }
  for (int o = 16; o > 0; o >>= 1) v += __shfl_down(v, o);
  if ((threadIdx.x & 31) == 0) sred[threadIdx.x >> 5] = v;
  __syncthreads();
  const float rstd = rsqrtf((sred[0] + sred[1] + sred[2] + sred[3]) / d + eps);
  for (int i = threadIdx.x; i < d; i += blockDim.x)
    yr[i] = (xr[i] - mean) * rstd * g[i] + b[i];
}

// ------------------------------ Softmax ------------------------------------
// softmax over `count` dim: elem(o,c,i) at x[(o*count + c)*inner + i]
__global__ void softmax_kernel(const float* __restrict__ x, float* __restrict__ y,
                               long outer, int count, long inner) {
  const long t = (long)blockIdx.x * blockDim.x + threadIdx.x;
  if (t >= outer * inner) return;
  const long o = t / inner, i = t % inner;
  const float* p = x + o * count * inner + i;
  float* q = y + o * count * inner + i;
  float mx = -3.0e38f;
  for (int c = 0; c < count; ++c) mx = fmaxf(mx, p[(long)c * inner]);
  float s = 0.f;
  for (int c = 0; c < count; ++c) s += __expf(p[(long)c * inner] - mx);
  const float inv = 1.f / s;
  for (int c = 0; c < count; ++c) q[(long)c * inner] = __expf(p[(long)c * inner] - mx) * inv;
}

// --------------------------- small helper kernels --------------------------
__global__ void seqw_dot_kernel(const float* q, const float* k, float* s,
                                int Nn, int L, int d) {
  const int t = blockIdx.x * blockDim.x + threadIdx.x;
  if (t >= Nn * L) return;
  const int n = t / L, l = t % L;
  const float* qp = q + (long)l * d;
  const float* kp = k + ((long)n * L + l) * d;
  float acc = 0.f;
  for (int i = 0; i < d; ++i) acc += qp[i] * kp[i];
  s[t] = acc;
}
__global__ void rowmul_kernel(const float* x, const float* w, float* y, long total, int P) {
  const long t = (long)blockIdx.x * blockDim.x + threadIdx.x;
  if (t >= total) return;
  y[t] = x[t] * w[t / P];
}
__global__ void sumn_kernel(const float* x, float* y, int n, long stride, long total) {
  const long t = (long)blockIdx.x * blockDim.x + threadIdx.x;
  if (t >= total) return;
  float s = 0.f;
  for (int i = 0; i < n; ++i) s += x[(long)i * stride + t];
  y[t] = s;
}
__global__ void feat_kernel(const float* f1s, const float* xd0, float* feat, int L, int P) {
  const int t = blockIdx.x * blockDim.x + threadIdx.x;
  if (t >= L * 2 * P) return;
  const int l = t / (2 * P), c = t % (2 * P);
  feat[t] = (c < P) ? f1s[l * P + c] : xd0[(long)l * P + (c - P)];
}
__global__ void convin_kernel(const float* po, const float* pn, const float* feat,
                              const float* att, float* out, int L) {
  const long t = (long)blockIdx.x * blockDim.x + threadIdx.x;
  const long total = (long)L * L * 388;
  if (t >= total) return;
  const long ij = t / 388; const int c = (int)(t % 388);
  const int i = (int)(ij / L), j = (int)(ij % L);
  float v;
  if (c < 128)      v = po[ij * 128 + c];
  else if (c < 256) v = pn[ij * 128 + (c - 128)];
  else if (c < 320) v = feat[(long)i * 64 + (c - 256)];
  else if (c < 384) v = feat[(long)j * 64 + (c - 320)];
  else              v = att[(long)(c - 384) * L * L + ij];
  out[t] = v;
}
__global__ void wsum_kernel(const float* w, const float* m, float* out, int Nn, int L, int d) {
  const int t = blockIdx.x * blockDim.x + threadIdx.x;
  if (t >= L * d) return;
  const int l = t / d, c = t % d;
  float s = 0.f;
  for (int n = 0; n < Nn; ++n) s += w[n * L + l] * m[((long)n * L + l) * d + c];
  out[t] = s;
}
__global__ void nodecat_kernel(const float* m1, const float* oh, float* out, int L) {
  const int t = blockIdx.x * blockDim.x + threadIdx.x;
  if (t >= L * 85) return;
  const int l = t / 85, c = t % 85;
  out[t] = (c < 64) ? m1[l * 64 + c] : oh[l * 21 + (c - 64)];
}
__global__ void topk_kernel(const float* __restrict__ xyz, const int* __restrict__ idx,
                            float* __restrict__ mask, float* __restrict__ degv,
                            int L, int kk, float kmin) {
  __shared__ float dd[256];
  __shared__ float mk[256];
  const int i = blockIdx.x;
  for (int j = threadIdx.x; j < L; j += blockDim.x) {
    const float d0 = xyz[j * 9 + 3] - xyz[i * 9 + 3];
    const float d1 = xyz[j * 9 + 4] - xyz[i * 9 + 4];
    const float d2 = xyz[j * 9 + 5] - xyz[i * 9 + 5];
    const float e = (i == j) ? 999.9f : 0.f;
    dd[j] = sqrtf(d0 * d0 + d1 * d1 + d2 * d2 + 1e-12f) + e;
    const float sep = fabsf((float)(idx[j] - idx[i])) + e;
    mk[j] = (sep < kmin) ? 1.f : 0.f;
  }
  __syncthreads();
  if (threadIdx.x == 0) {
    for (int t = 0; t < kk; ++t) {
      int am = 0; float bvv = 3.0e38f;
      for (int j = 0; j < L; ++j) if (dd[j] < bvv) { bvv = dd[j]; am = j; }
      mk[am] = 1.f; dd[am] = 3.0e38f;
    }
    float s = 0.f;
    for (int j = 0; j < L; ++j) s += mk[j];
    degv[i] = fmaxf(s, 1.f);
  }
  __syncthreads();
  for (int j = threadIdx.x; j < L; j += blockDim.x) mask[(long)i * L + j] = mk[j];
}
__global__ void minp_kernel(const float* __restrict__ xyz, const float* __restrict__ node,
                            const float* __restrict__ edge, float* __restrict__ minp,
                            float* __restrict__ dhat, int L) {
  const long t = (long)blockIdx.x * blockDim.x + threadIdx.x;
  if (t >= (long)L * L) return;
  const int i = (int)(t / L), j = (int)(t % L);
  const float dv0 = xyz[j * 9 + 3] - xyz[i * 9 + 3];
  const float dv1 = xyz[j * 9 + 4] - xyz[i * 9 + 4];
  const float dv2 = xyz[j * 9 + 5] - xyz[i * 9 + 5];
  const float dn = sqrtf(dv0 * dv0 + dv1 * dv1 + dv2 * dv2 + 1e-8f);
  const float inv = 1.f / dn;
  dhat[t * 3 + 0] = dv0 * inv; dhat[t * 3 + 1] = dv1 * inv; dhat[t * 3 + 2] = dv2 * inv;
  float* mp = minp + t * 97;
  const float* ni = node + (long)i * 32;
  const float* nj = node + (long)j * 32;
  const float* ed = edge + t * 32;
  for (int c = 0; c < 32; ++c) { mp[c] = ni[c]; mp[32 + c] = nj[c]; mp[64 + c] = ed[c]; }
  mp[96] = dn;
}
__global__ void statepre_kernel(const float* msg, const float* mask, const float* degv,
                                float* sp, int L) {
  const int t = blockIdx.x * blockDim.x + threadIdx.x;
  if (t >= L * 32) return;
  const int i = t / 32, c = t % 32;
  float s = 0.f;
  for (int j = 0; j < L; ++j)
    s += msg[((long)i * L + j) * 44 + c] * mask[(long)i * L + j];
  sp[t] = s / degv[i];
}
__global__ void aggs_kernel(const float* __restrict__ msg, const float* __restrict__ mask,
                            const float* __restrict__ dhat, const float* __restrict__ xyz,
                            const float* __restrict__ degv, const float* __restrict__ Wself,
                            float* __restrict__ offset, int L) {
  const int t = blockIdx.x * blockDim.x + threadIdx.x;
  if (t >= L * 9) return;
  const int i = t / 9, r = t % 9, o = r / 3, x = r % 3;
  float s = 0.f;
  for (int j = 0; j < L; ++j) {
    const float mk = mask[(long)i * L + j];
    if (mk == 0.f) continue;
    const float* mg = msg + ((long)i * L + j) * 44;
    float a = mg[32 + o] * dhat[((long)i * L + j) * 3 + x];
    float b = 0.f;
    for (int c = 0; c < 3; ++c)
      b += mg[35 + o * 3 + c] * (xyz[j * 9 + c * 3 + x] - xyz[j * 9 + 3 + x]);
    s += mk * (a + b);
  }
  s /= degv[i];
  float self = 0.f;
  for (int c = 0; c < 3; ++c)
    self += Wself[o * 3 + c] * (xyz[i * 9 + c * 3 + x] - xyz[i * 9 + 3 + x]);
  offset[t] = s + self;
}
__global__ void coords_kernel(const float* xyz, const float* off, float* xyz_out,
                              float* caN, int L) {
  const int i = blockIdx.x * blockDim.x + threadIdx.x;
  if (i >= L) return;
  for (int x = 0; x < 3; ++x) {
    const float ca = xyz[i * 9 + 3 + x];
    const float can = ca + off[i * 9 + 3 + x];           // o=1 row
    xyz_out[i * 9 + 0 + x] = can + off[i * 9 + 0 + x];   // N
    xyz_out[i * 9 + 3 + x] = can;                        // CA
    xyz_out[i * 9 + 6 + x] = can + off[i * 9 + 6 + x];   // C
    caN[i * 3 + x] = can;
  }
}
__global__ void masks_kernel(const float* caN, float* ms, int L) {
  const long t = (long)blockIdx.x * blockDim.x + threadIdx.x;
  const long LL2 = (long)L * L;
  if (t >= 4 * LL2) return;
  const int h = (int)(t / LL2);
  const long ij = t % LL2;
  const int i = (int)(ij / L), j = (int)(ij % L);
  const float d0 = caN[i * 3 + 0] - caN[j * 3 + 0];
  const float d1 = caN[i * 3 + 1] - caN[j * 3 + 1];
  const float d2 = caN[i * 3 + 2] - caN[j * 3 + 2];
  const float dist = sqrtf(d0 * d0 + d1 * d1 + d2 * d2 + 1e-12f);
  const float bins[4] = {8.f, 12.f, 16.f, 20.f};
  ms[t] = 1.f - 1.f / (1.f + __expf(-(dist - bins[h])));
}
__global__ void mulmask_kernel(float* a, const float* m, long total) {
  const long t = (long)blockIdx.x * blockDim.x + threadIdx.x;
  if (t < total) a[t] *= m[t];
}

// ------------------------------- host side ---------------------------------
static inline GemmP gp0() {
  GemmP g; memset(&g, 0, sizeof(g));
  g.K2 = 1; g.nb2 = 1; g.MiC = 1; g.NiC = 1; g.alpha = 1.f;
  return g;
}

extern "C" void kernel_launch(void* const* d_in, const int* in_sizes, int n_in,
                              void* d_out, int out_size, void* d_ws, size_t ws_size,
                              hipStream_t stream) {
  (void)in_sizes; (void)n_in; (void)out_size; (void)ws_size;
  const int L = 192, NS = 32;
  const long NL = (long)NS * L;        // 6144
  const long LL = (long)L * L;         // 36864

  const float* msa_in  = (const float*)d_in[0];
  const float* pair_in = (const float*)d_in[1];
  const float* xyz_in  = (const float*)d_in[2];
  const float* seq1hot = (const float*)d_in[3];
  const int*   idx_in  = (const int*)d_in[4];
  // d_in[5] = top_k (static 64)

  int pi = 6;
  auto nx = [&]() { return (const float*)d_in[pi++]; };
  // ---- params in make_params() insertion order ----
  const float *e1_ln1g=nx(),*e1_ln1b=nx(),*e1_Wq=nx(),*e1_bq=nx(),*e1_Wk=nx(),*e1_bk=nx(),
              *e1_Wv=nx(),*e1_bv=nx(),*e1_Wo=nx(),*e1_bo=nx(),*e1_ln2g=nx(),*e1_ln2b=nx(),
              *e1_fW1=nx(),*e1_fb1=nx(),*e1_fW2=nx(),*e1_fb2=nx();
  const float *e2_ln1g=nx(),*e2_ln1b=nx(),*e2_Wq=nx(),*e2_bq=nx(),*e2_Wk=nx(),*e2_bk=nx(),
              *e2_Wv=nx(),*e2_bv=nx(),*e2_Wo=nx(),*e2_bo=nx(),*e2_ln2g=nx(),*e2_ln2b=nx(),
              *e2_fW1=nx(),*e2_fb1=nx(),*e2_fW2=nx(),*e2_fb2=nx();
  const float *mp_n1g=nx(),*mp_n1b=nx(),*mp_pW=nx(),*mp_pb=nx(),*mp_ndg=nx(),*mp_ndb=nx(),
              *mp_swq=nx(),*mp_swk=nx(),*mp_n2g=nx(),*mp_n2b=nx(),*mp_cW=nx(),*mp_cb=nx(),
              *mp_nog=nx(),*mp_nob=nx(),*mp_nng=nx(),*mp_nnb=nx(),*mp_ci=nx(),*mp_cib=nx(),
              *mp_c1=nx(),*mp_c1b=nx(),*mp_c2=nx(),*mp_c2b=nx(),*mp_co=nx(),*mp_cob=nx();
  const float *pp_ln1g=nx(),*pp_ln1b=nx(),
              *pp_rWq=nx(),*pp_rbq=nx(),*pp_rWk=nx(),*pp_rbk=nx(),*pp_rWv=nx(),*pp_rbv=nx(),
              *pp_rWo=nx(),*pp_rbo=nx(),
              *pp_cWq=nx(),*pp_cbq=nx(),*pp_cWk=nx(),*pp_cbk=nx(),*pp_cWv=nx(),*pp_cbv=nx(),
              *pp_cWo=nx(),*pp_cbo=nx(),
              *pp_ln2g=nx(),*pp_ln2b=nx(),*pp_ln3g=nx(),*pp_ln3b=nx(),
              *pp_fW1=nx(),*pp_fb1=nx(),*pp_fW2=nx(),*pp_fb2=nx();
  const float *pm_lpg=nx(),*pm_lpb=nx(),*pm_Wl=nx(),*pm_bl=nx(),*pm_lmg=nx(),*pm_lmb=nx(),
              *pm_Wv=nx(),*pm_bv=nx(),*pm_Wo=nx(),*pm_bo=nx(),*pm_ln2g=nx(),*pm_ln2b=nx(),
              *pm_fW1=nx(),*pm_fb1=nx(),*pm_fW2=nx(),*pm_fb2=nx();
  const float *ss_nmg=nx(),*ss_nmb=nx(),*ss_npg=nx(),*ss_npb=nx(),*ss_swq=nx(),*ss_swk=nx(),
              *ss_exW=nx(),*ss_exb=nx(),*ss_eeW=nx(),*ss_eeb=nx(),*ss_nng=nx(),*ss_nnb=nx(),
              *ss_neg=nx(),*ss_neb=nx(),*ss_m1=nx(),*ss_m1b=nx(),*ss_m2=nx(),*ss_m2b=nx(),
              *ss_Ws=nx(),*ss_bs=nx(),*ss_Wself=nx();
  const float *sm_nsg=nx(),*sm_nsb=nx(),*sm_n1g=nx(),*sm_n1b=nx(),*sm_Wq=nx(),*sm_Wk=nx(),
              *sm_Wv=nx(),*sm_Wo=nx(),*sm_bo=nx(),*sm_n2g=nx(),*sm_n2b=nx(),
              *sm_fW1=nx(),*sm_fb1=nx(),*sm_fW2=nx(),*sm_fb2=nx();

  // ---- outputs ----
  float* out_msa  = (float*)d_out;                 // 1*32*192*64
  float* out_pair = out_msa + NL * 64;             // 1*192*192*128
  float* out_xyz  = out_pair + LL * 128;           // 1*192*3*3

  // ---- workspace bump allocator ----
  char* wsp = (char*)d_ws;
  size_t woff = 0;
  auto FB = [&](long n) {
    float* p = (float*)(wsp + woff);
    woff += (((size_t)n * 4) + 255) & ~(size_t)255;
    return p;
  };
  float *lnM = FB(NL*64), *lnP = FB(LL*128);
  float *bufQ = FB(NL*64), *bufK = FB(NL*64), *bufV = FB(NL*64), *bufO = FB(NL*64);
  float *ffnH = FB(NL*256);
  float *msaA = FB(NL*64), *msaB = FB(NL*64), *msaC = FB(NL*64), *msaD = FB(NL*64),
        *msaE = FB(NL*64), *msaF = FB(NL*64), *msaG = FB(NL*64);
  float *attT = FB(4*LL);
  float *scores = FB((long)L*8*LL);                // 56.6M f, reused by all attn
  float *xd0 = FB(NL*32), *xd = FB(NL*32);
  float *qsB = FB((long)L*64), *ksW = FB(NL*64), *sWt = FB((long)NS*L);
  float *f1 = FB(NL*32), *f1s = FB((long)L*32), *featB = FB((long)L*64);
  float *outerB = FB(LL*1024);
  float *pnew = FB(LL*128), *poB = FB(LL*128);
  float *convin = FB(LL*388);
  float *convX = FB(LL*128), *convH = FB(LL*128), *convY = FB(LL*128), *pairA = FB(LL*128);
  float *pQ = FB(LL*128), *pK = FB(LL*128), *pV = FB(LL*128), *pO = FB(LL*128);
  float *pairB = FB(LL*128), *pairC = FB(LL*128), *pffnH = FB(LL*512);
  float *lgB = FB(LL*4), *aPM = FB(LL*4);
  float *m1B = FB((long)L*64), *ncat = FB((long)L*85), *nodeB = FB((long)L*32);
  float *edgeB = FB(LL*32), *maskB = FB(LL), *degv = FB(L);
  float *minpB = FB(LL*97), *dhatB = FB(LL*3);
  float *h32 = FB(LL*32), *msgB = FB(LL*44);
  float *spre = FB((long)L*32), *stateB = FB((long)L*16), *offs = FB((long)L*9), *caN = FB((long)L*3);
  float *maskS = FB(4*LL), *stB = FB((long)L*16), *sq = FB((long)L*128), *sk = FB((long)L*128);
  float *aS = FB(4*LL), *sv = FB(NL*128), *so = FB(NL*128);

  // ---- launch helpers ----
  auto GEMM = [&](const GemmP& g, int nb1) {
    dim3 gr((unsigned)((g.M + 31) / 32), (unsigned)((g.N + 63) / 64),
            (unsigned)(nb1 * g.nb2));
    gemm_wmma_kernel<<<gr, 128, 0, stream>>>(g);
  };
  auto LIN = [&](const float* A, const float* W, const float* bias, const float* Cin,
                 float* C, int rows, int in, int out, int act, float alpha, int preact) {
    GemmP g = gp0();
    g.A = A; g.B = W; g.bias = bias; g.Cin = Cin; g.C = C;
    g.M = rows; g.N = out; g.K = in;
    g.rsA = in; g.ksA = 1; g.csB = 1; g.ksB = out;
    g.rsC = out; g.csC = 1;
    g.act = act; g.alpha = alpha; g.preact = preact;
    GEMM(g, 1);
  };
  auto LN = [&](const float* x, const float* g_, const float* b_, float* y, long rows, int d) {
    ln_kernel<<<(unsigned)rows, 128, 0, stream>>>(x, g_, b_, y, d, 1e-5f);
  };
  auto SMAX = [&](const float* x, float* y, long outer, int count, long inner) {
    const long rows = outer * inner;
    softmax_kernel<<<(unsigned)((rows + 255) / 256), 256, 0, stream>>>(x, y, outer, count, inner);
  };
  auto EW = [&](long total) { return (unsigned)((total + 255) / 256); };

  // ======================= E1: tied MSA attention ==========================
  LN(msa_in, e1_ln1g, e1_ln1b, lnM, NL, 64);
  LIN(lnM, e1_Wq, e1_bq, nullptr, bufQ, (int)NL, 64, 64, 0, 1.f, 0);
  LIN(lnM, e1_Wk, e1_bk, nullptr, bufK, (int)NL, 64, 64, 0, 1.f, 0);
  LIN(lnM, e1_Wv, e1_bv, nullptr, bufV, (int)NL, 64, 64, 0, 1.f, 0);
  { GemmP g = gp0(); g.A=bufQ; g.B=bufK; g.C=scores;
    g.M=L; g.N=L; g.K=16; g.K2=NS; g.nb2=4;
    g.kshift=4; g.kmask=15;                         // K=16 -> shift/mask
    g.rsA=64; g.ksA=1; g.k2sA=12288; g.b2A=16;
    g.csB=64; g.ksB=1; g.k2sB=12288; g.b2B=16;
    g.rsC=L; g.csC=1; g.b2C=LL;
    g.alpha=0.04419417382f;                         // 1/sqrt(16*32)
    GEMM(g, 1); }
  SMAX(scores, attT, 4L*L, L, 1);
  { GemmP g = gp0(); g.A=attT; g.B=bufV; g.C=bufO;
    g.M=L; g.N=16; g.K=L; g.nb2=4;
    g.rsA=L; g.ksA=1; g.b2A=LL;
    g.ksB=64; g.csB=1; g.b1B=12288; g.b2B=16;
    g.rsC=64; g.csC=1; g.b1C=12288; g.b2C=16;
    GEMM(g, NS); }
  LIN(bufO, e1_Wo, e1_bo, msa_in, msaA, (int)NL, 64, 64, 0, 1.f, 0);
  LN(msaA, e1_ln2g, e1_ln2b, lnM, NL, 64);
  LIN(lnM, e1_fW1, e1_fb1, nullptr, ffnH, (int)NL, 64, 256, 1, 1.f, 0);
  LIN(ffnH, e1_fW2, e1_fb2, msaA, msaB, (int)NL, 256, 64, 0, 1.f, 0);

  // ================== E2: attention over the N dimension ===================
  LN(msaB, e2_ln1g, e2_ln1b, lnM, NL, 64);
  LIN(lnM, e2_Wq, e2_bq, nullptr, bufQ, (int)NL, 64, 64, 0, 1.f, 0);
  LIN(lnM, e2_Wk, e2_bk, nullptr, bufK, (int)NL, 64, 64, 0, 1.f, 0);
  LIN(lnM, e2_Wv, e2_bv, nullptr, bufV, (int)NL, 64, 64, 0, 1.f, 0);
  { GemmP g = gp0(); g.A=bufQ; g.B=bufK; g.C=scores;
    g.M=NS; g.N=NS; g.K=16; g.nb2=4;
    g.rsA=12288; g.ksA=1; g.b1A=64; g.b2A=16;
    g.csB=12288; g.ksB=1; g.b1B=64; g.b2B=16;
    g.rsC=NS; g.csC=1; g.b1C=4096; g.b2C=1024;
    g.alpha=0.25f; GEMM(g, L); }
  SMAX(scores, scores, (long)L*4*NS, NS, 1);
  { GemmP g = gp0(); g.A=scores; g.B=bufV; g.C=bufO;
    g.M=NS; g.N=16; g.K=NS; g.nb2=4;
    g.rsA=NS; g.ksA=1; g.b1A=4096; g.b2A=1024;
    g.ksB=12288; g.csB=1; g.b1B=64; g.b2B=16;
    g.rsC=12288; g.csC=1; g.b1C=64; g.b2C=16;
    GEMM(g, L); }
  LIN(bufO, e2_Wo, e2_bo, msaB, msaC, (int)NL, 64, 64, 0, 1.f, 0);
  LN(msaC, e2_ln2g, e2_ln2b, lnM, NL, 64);
  LIN(lnM, e2_fW1, e2_fb1, nullptr, ffnH, (int)NL, 64, 256, 1, 1.f, 0);
  LIN(ffnH, e2_fW2, e2_fb2, msaC, msaD, (int)NL, 256, 64, 0, 1.f, 0);

  // ============== MP: outer product pair update + conv resnet ==============
  LN(msaD, mp_n1g, mp_n1b, lnM, NL, 64);
  LIN(lnM, mp_pW, mp_pb, nullptr, xd0, (int)NL, 64, 32, 0, 1.f, 0);
  LN(xd0, mp_ndg, mp_ndb, xd, NL, 32);
  LIN(xd, mp_swq, nullptr, nullptr, qsB, L, 32, 32, 0, 0.1767766953f, 0);
  LIN(xd, mp_swk, nullptr, nullptr, ksW, (int)NL, 32, 32, 0, 1.f, 0);
  seqw_dot_kernel<<<EW(NS*L), 256, 0, stream>>>(qsB, ksW, sWt, NS, L, 32);
  SMAX(sWt, sWt, 1, NS, L);
  rowmul_kernel<<<EW(NL*32), 256, 0, stream>>>(xd, sWt, f1, NL*32, 32);
  { GemmP g = gp0(); g.A=xd; g.B=f1; g.C=outerB;           // bnip,bnjq->bijpq
    g.M=(int)(L*32); g.N=(int)(L*32); g.K=NS;
    g.rsA=1; g.ksA=6144; g.csB=1; g.ksB=6144;
    g.MiC=32; g.rsC=(long)L*1024; g.rsC2=32;
    g.NiC=32; g.csC=1024; g.csC2=1;
    GEMM(g, 1); }
  LN(outerB, mp_n2g, mp_n2b, outerB, LL, 1024);
  LIN(outerB, mp_cW, mp_cb, nullptr, pnew, (int)LL, 1024, 128, 0, 1.f, 0);
  sumn_kernel<<<EW(6144), 256, 0, stream>>>(f1, f1s, NS, 6144, 6144);
  feat_kernel<<<EW(L*64), 256, 0, stream>>>(f1s, xd, featB, L, 32);
  LN(pair_in, mp_nog, mp_nob, poB, LL, 128);
  LN(pnew, mp_nng, mp_nnb, pnew, LL, 128);
  convin_kernel<<<EW(LL*388), 256, 0, stream>>>(poB, pnew, featB, attT, convin, L);
  { GemmP g = gp0(); g.A=convin; g.B=mp_ci; g.bias=mp_cib; g.C=convX;   // 1x1, OIHW
    g.M=(int)LL; g.N=128; g.K=388;
    g.rsA=388; g.ksA=1; g.csB=388; g.ksB=1; g.rsC=128; g.csC=1;
    GEMM(g, 1); }
  { ConvP c; c.in=convX; c.W=mp_c1; c.bias=mp_c1b; c.Cin=nullptr; c.out=convH;
    c.H=L; c.Wd=L; c.Ci=128; c.Co=128; c.preelu=1;
    dim3 gr((unsigned)((LL+31)/32), 2, 1);
    conv3x3_wmma_kernel<<<gr, 128, 0, stream>>>(c); }
  { ConvP c; c.in=convH; c.W=mp_c2; c.bias=mp_c2b; c.Cin=convX; c.out=convY;
    c.H=L; c.Wd=L; c.Ci=128; c.Co=128; c.preelu=1;
    dim3 gr((unsigned)((LL+31)/32), 2, 1);
    conv3x3_wmma_kernel<<<gr, 128, 0, stream>>>(c); }
  { GemmP g = gp0(); g.A=convY; g.B=mp_co; g.bias=mp_cob; g.C=pairA;    // 1x1 + pre-ELU
    g.M=(int)LL; g.N=128; g.K=128; g.preact=2;
    g.rsA=128; g.ksA=1; g.csB=128; g.ksB=1; g.rsC=128; g.csC=1;
    GEMM(g, 1); }

  // ===================== PP: pair row / column attention ===================
  LN(pairA, pp_ln1g, pp_ln1b, lnP, LL, 128);
  LIN(lnP, pp_rWq, pp_rbq, nullptr, pQ, (int)LL, 128, 128, 0, 1.f, 0);
  LIN(lnP, pp_rWk, pp_rbk, nullptr, pK, (int)LL, 128, 128, 0, 1.f, 0);
  LIN(lnP, pp_rWv, pp_rbv, nullptr, pV, (int)LL, 128, 128, 0, 1.f, 0);
  { GemmP g = gp0(); g.A=pQ; g.B=pK; g.C=scores;
    g.M=L; g.N=L; g.K=16; g.nb2=8;
    g.rsA=128; g.ksA=1; g.b1A=24576; g.b2A=16;
    g.csB=128; g.ksB=1; g.b1B=24576; g.b2B=16;
    g.rsC=L; g.csC=1; g.b1C=8*LL; g.b2C=LL;
    g.alpha=0.25f; GEMM(g, L); }
  SMAX(scores, scores, (long)L*8*L, L, 1);
  { GemmP g = gp0(); g.A=scores; g.B=pV; g.C=pO;
    g.M=L; g.N=16; g.K=L; g.nb2=8;
    g.rsA=L; g.ksA=1; g.b1A=8*LL; g.b2A=LL;
    g.ksB=128; g.csB=1; g.b1B=24576; g.b2B=16;
    g.rsC=128; g.csC=1; g.b1C=24576; g.b2C=16;
    GEMM(g, L); }
  LIN(pO, pp_rWo, pp_rbo, pairA, pairB, (int)LL, 128, 128, 0, 1.f, 0);
  LN(pairB, pp_ln2g, pp_ln2b, lnP, LL, 128);
  LIN(lnP, pp_cWq, pp_cbq, nullptr, pQ, (int)LL, 128, 128, 0, 1.f, 0);
  LIN(lnP, pp_cWk, pp_cbk, nullptr, pK, (int)LL, 128, 128, 0, 1.f, 0);
  LIN(lnP, pp_cWv, pp_cbv, nullptr, pV, (int)LL, 128, 128, 0, 1.f, 0);
  { GemmP g = gp0(); g.A=pQ; g.B=pK; g.C=scores;        // attention over i, fixed j
    g.M=L; g.N=L; g.K=16; g.nb2=8;
    g.rsA=24576; g.ksA=1; g.b1A=128; g.b2A=16;
    g.csB=24576; g.ksB=1; g.b1B=128; g.b2B=16;
    g.rsC=L; g.csC=1; g.b1C=8*LL; g.b2C=LL;
    g.alpha=0.25f; GEMM(g, L); }
  SMAX(scores, scores, (long)L*8*L, L, 1);
  { GemmP g = gp0(); g.A=scores; g.B=pV; g.C=pO;
    g.M=L; g.N=16; g.K=L; g.nb2=8;
    g.rsA=L; g.ksA=1; g.b1A=8*LL; g.b2A=LL;
    g.ksB=24576; g.csB=1; g.b1B=128; g.b2B=16;
    g.rsC=24576; g.csC=1; g.b1C=128; g.b2C=16;
    GEMM(g, L); }
  LIN(pO, pp_cWo, pp_cbo, pairB, pairC, (int)LL, 128, 128, 0, 1.f, 0);
  LN(pairC, pp_ln3g, pp_ln3b, lnP, LL, 128);
  LIN(lnP, pp_fW1, pp_fb1, nullptr, pffnH, (int)LL, 128, 512, 1, 1.f, 0);
  LIN(pffnH, pp_fW2, pp_fb2, pairC, out_pair, (int)LL, 512, 128, 0, 1.f, 0);  // final pair

  // ===================== PM: pair -> MSA gated update ======================
  LN(out_pair, pm_lpg, pm_lpb, lnP, LL, 128);
  LIN(lnP, pm_Wl, pm_bl, nullptr, lgB, (int)LL, 128, 4, 0, 1.f, 0);
  SMAX(lgB, aPM, L, L, 4);
  LN(msaD, pm_lmg, pm_lmb, lnM, NL, 64);
  LIN(lnM, pm_Wv, pm_bv, nullptr, bufV, (int)NL, 64, 64, 0, 1.f, 0);
  { GemmP g = gp0(); g.A=aPM; g.B=bufV; g.C=bufO;
    g.M=L; g.N=16; g.K=L; g.nb2=4;
    g.rsA=768; g.ksA=4; g.b2A=1;
    g.ksB=64; g.csB=1; g.b1B=12288; g.b2B=16;
    g.rsC=64; g.csC=1; g.b1C=12288; g.b2C=16;
    GEMM(g, NS); }
  LIN(bufO, pm_Wo, pm_bo, msaD, msaE, (int)NL, 64, 64, 0, 1.f, 0);
  LN(msaE, pm_ln2g, pm_ln2b, lnM, NL, 64);
  LIN(lnM, pm_fW1, pm_fb1, nullptr, ffnH, (int)NL, 64, 256, 1, 1.f, 0);
  LIN(ffnH, pm_fW2, pm_fb2, msaE, msaF, (int)NL, 256, 64, 0, 1.f, 0);

  // ================== SS: graph message passing / structure ================
  LN(msaF, ss_nmg, ss_nmb, lnM, NL, 64);
  LN(out_pair, ss_npg, ss_npb, lnP, LL, 128);
  LIN(lnM, ss_swq, nullptr, nullptr, qsB, L, 64, 64, 0, 0.125f, 0);
  LIN(lnM, ss_swk, nullptr, nullptr, ksW, (int)NL, 64, 64, 0, 1.f, 0);
  seqw_dot_kernel<<<EW(NS*L), 256, 0, stream>>>(qsB, ksW, sWt, NS, L, 64);
  SMAX(sWt, sWt, 1, NS, L);
  wsum_kernel<<<EW(L*64), 256, 0, stream>>>(sWt, lnM, m1B, NS, L, 64);
  nodecat_kernel<<<EW(L*85), 256, 0, stream>>>(m1B, seq1hot, ncat, L);
  LIN(ncat, ss_exW, ss_exb, nullptr, nodeB, L, 85, 32, 0, 1.f, 0);
  LN(nodeB, ss_nng, ss_nnb, nodeB, L, 32);
  LIN(lnP, ss_eeW, ss_eeb, nullptr, edgeB, (int)LL, 128, 32, 0, 1.f, 0);
  LN(edgeB, ss_neg, ss_neb, edgeB, LL, 32);
  topk_kernel<<<L, 256, 0, stream>>>(xyz_in, idx_in, maskB, degv, L, 64, 9.0f);
  minp_kernel<<<EW(LL), 256, 0, stream>>>(xyz_in, nodeB, edgeB, minpB, dhatB, L);
  LIN(minpB, ss_m1, ss_m1b, nullptr, h32, (int)LL, 97, 32, 1, 1.f, 0);
  LIN(h32, ss_m2, ss_m2b, nullptr, msgB, (int)LL, 32, 44, 0, 1.f, 0);
  statepre_kernel<<<EW(L*32), 256, 0, stream>>>(msgB, maskB, degv, spre, L);
  LIN(spre, ss_Ws, ss_bs, nullptr, stateB, L, 32, 16, 0, 1.f, 0);
  aggs_kernel<<<EW(L*9), 256, 0, stream>>>(msgB, maskB, dhatB, xyz_in, degv, ss_Wself, offs, L);
  coords_kernel<<<1, 256, 0, stream>>>(xyz_in, offs, out_xyz, caN, L);   // final xyz
  masks_kernel<<<EW(4*LL), 256, 0, stream>>>(caN, maskS, L);

  // =============== SM: structure-biased attention + final FFN ==============
  LN(stateB, sm_nsg, sm_nsb, stB, L, 16);
  LN(msaF, sm_n1g, sm_n1b, lnM, NL, 64);
  LIN(stB, sm_Wq, nullptr, nullptr, sq, L, 16, 128, 0, 1.f, 0);
  LIN(stB, sm_Wk, nullptr, nullptr, sk, L, 16, 128, 0, 1.f, 0);
  { GemmP g = gp0(); g.A=sq; g.B=sk; g.C=scores;
    g.M=L; g.N=L; g.K=32; g.nb2=4;
    g.rsA=128; g.ksA=1; g.b2A=32;
    g.csB=128; g.ksB=1; g.b2B=32;
    g.rsC=L; g.csC=1; g.b2C=LL;
    g.alpha=0.1767766953f; GEMM(g, 1); }
  SMAX(scores, aS, 4L*L, L, 1);
  mulmask_kernel<<<EW(4*LL), 256, 0, stream>>>(aS, maskS, 4*LL);
  LIN(lnM, sm_Wv, nullptr, nullptr, sv, (int)NL, 64, 128, 0, 1.f, 0);
  { GemmP g = gp0(); g.A=aS; g.B=sv; g.C=so;
    g.M=L; g.N=32; g.K=L; g.nb2=4;
    g.rsA=L; g.ksA=1; g.b2A=LL;
    g.ksB=128; g.csB=1; g.b1B=24576; g.b2B=32;
    g.rsC=128; g.csC=1; g.b1C=24576; g.b2C=32;
    GEMM(g, NS); }
  LIN(so, sm_Wo, sm_bo, msaF, msaG, (int)NL, 128, 64, 0, 1.f, 0);
  LN(msaG, sm_n2g, sm_n2b, lnM, NL, 64);
  LIN(lnM, sm_fW1, sm_fb1, nullptr, ffnH, (int)NL, 64, 256, 1, 1.f, 0);
  LIN(ffnH, sm_fW2, sm_fb2, msaG, out_msa, (int)NL, 256, 64, 0, 1.f, 0);   // final msa
}
